// SourceTargetHadamardMLPRoute_29669634081461
// MI455X (gfx1250) — compile-verified, matrix operand order checked by reading, not run on hardware
//
#include <hip/hip_runtime.h>

typedef __attribute__((ext_vector_type(16))) _Float16 v16h;
typedef __attribute__((ext_vector_type(2)))  _Float16 h2;
typedef __attribute__((ext_vector_type(8)))  float    v8f;

#define N_SRC   1024
#define N_TGT   1024
#define SRC_DIM 256
#define DST_DIM 256
#define WIDTH   256
#define HIDDEN  128
#define HWPAD   264   // padded row length in halves: 528B stride -> conflict-free

union Frag { v16h v; h2 p[8]; };

// ---------------------------------------------------------------------------
// Stage 1a: P = X @ W^T + b   (f32 + f16 copy).  1024 blocks x 256 threads.
// ---------------------------------------------------------------------------
__global__ void proj_kernel(const float* __restrict__ X,
                            const float* __restrict__ W,
                            const float* __restrict__ b,
                            float* __restrict__ Pf32,
                            _Float16* __restrict__ Pf16)
{
    const int n = blockIdx.x;
    const int t = threadIdx.x;                 // output column, 0..255
    const float* __restrict__ x = X + n * SRC_DIM;
    const float* __restrict__ w = W + t * SRC_DIM;
    float acc = 0.f;
    #pragma unroll 4
    for (int d = 0; d < SRC_DIM; d += 4) {
        acc += x[d + 0] * w[d + 0];
        acc += x[d + 1] * w[d + 1];
        acc += x[d + 2] * w[d + 2];
        acc += x[d + 3] * w[d + 3];
    }
    acc += b[t];
    Pf32[n * WIDTH + t] = acc;
    Pf16[n * WIDTH + t] = (_Float16)acc;
}

// ---------------------------------------------------------------------------
// Stage 1b: L[n,h] = P[n,:] . Wi[h, colOff:colOff+WIDTH] (+ bi[h])
// ---------------------------------------------------------------------------
__global__ void lin_kernel(const float* __restrict__ P,
                           const float* __restrict__ Wi,
                           const float* __restrict__ bi,
                           float* __restrict__ L,
                           int colOff, int useBias)
{
    const int n = blockIdx.x;
    const int h = threadIdx.x;                 // 0..127
    const float* __restrict__ p = P + n * WIDTH;
    const float* __restrict__ w = Wi + h * (3 * WIDTH) + colOff;
    float acc = useBias ? bi[h] : 0.f;
    #pragma unroll 4
    for (int d = 0; d < WIDTH; ++d) acc += p[d] * w[d];
    L[n * HIDDEN + h] = acc;
}

// ---------------------------------------------------------------------------
// Stage 1c: hw_f16[h][d] = (f16) Wi[h][2*WIDTH + d]
// ---------------------------------------------------------------------------
__global__ void cvt_hw_kernel(const float* __restrict__ Wi,
                              _Float16* __restrict__ hwH)
{
    const int i = blockIdx.x * 256 + threadIdx.x;   // 0 .. 128*256-1
    const int h = i >> 8;
    const int d = i & 255;
    hwH[i] = (_Float16)Wi[h * (3 * WIDTH) + 2 * WIDTH + d];
}

// ---------------------------------------------------------------------------
// Stage 2 (fused main kernel):
//   hidden[n,m,h] = sum_d hw[h,d]*(ps[n,d]*pt[m,d]) + srcLin[n,h] + tgtLin[m,h]
//   out[n,m]      = sum_h silu(hidden[n,m,h]) * Wo[h] + bo
// One wave per (n, 16-wide m tile); block = 8 waves (8 consecutive n).
// Two passes of 4 h-tiles; each pass stages its 64 hw rows in padded LDS so
// the inner loop is ds_load_b128 + v_wmma only.  The B fragment (ps .* pt)
// is ping-pong buffered and built one k-chunk ahead so its pk_muls never
// WAR-collide with registers a running WMMA is reading (no hazard NOPs).
// ---------------------------------------------------------------------------
__global__ __launch_bounds__(256) void route_kernel(
    const _Float16* __restrict__ psH,     // [1024][256]
    const _Float16* __restrict__ ptH,     // [1024][256]
    const _Float16* __restrict__ hwH,     // [128][256]
    const float*    __restrict__ srcLin,  // [1024][128]
    const float*    __restrict__ tgtLin,  // [1024][128]
    const float*    __restrict__ Wo,      // [128]
    const float*    __restrict__ bo,      // [1]
    float*          __restrict__ out)     // [1024][1024]
{
    __shared__ __align__(16) _Float16 sHW[64 * HWPAD]; // hw slab (one pass) 33KB
    __shared__ __align__(16) _Float16 sPT[16 * HWPAD]; // pt tile, padded 8.25KB
    __shared__ __align__(16) _Float16 sPS[8 * 256];    // ps rows 4KB

    const int mBase = blockIdx.x * 16;
    const int nBase = blockIdx.y * 8;
    const int tid   = threadIdx.x;

    // ---- cooperative tile fills (padded destination rows) ----
    {   // pt tile: thread t -> row t/16, 16 halves at col (t%16)*16
        const int row = tid >> 4, col = (tid & 15) * 16;
        const uint4* __restrict__ src =
            (const uint4*)(ptH + (mBase + row) * 256 + col);
        uint4* dst = (uint4*)(sPT + row * HWPAD + col);
        dst[0] = src[0];
        dst[1] = src[1];
    }
    {   // ps rows: thread t -> row t/32, 8 halves at col (t%32)*8
        const int row = tid >> 5, col = (tid & 31) * 8;
        *(uint4*)(sPS + row * 256 + col) =
            *(const uint4*)(psH + (nBase + row) * 256 + col);
    }

    const int wave = tid >> 5;
    const int lane = tid & 31;
    const int hsel = lane >> 4;      // which 16-lane half
    const int lrow = lane & 15;
    const int n    = nBase + wave;

    // K offsets inside a 32-wide chunk for the 16-bit A/B fragment layout:
    // lanes 0-15 : v0-3 -> K 0..7,  v4-7 -> K 16..23
    // lanes 16-31: v0-3 -> K 8..15, v4-7 -> K 24..31
    int kOff[8];
    #pragma unroll
    for (int v = 0; v < 8; ++v)
        kOff[v] = ((v < 4) ? 0 : 16) + (hsel ? 8 : 0) + 2 * (v & 3);

    const float* __restrict__ sl = srcLin + n * HIDDEN;
    const float* __restrict__ tl = tgtLin + (mBase + lrow) * HIDDEN;
    const _Float16* __restrict__ psRow = sPS + wave * 256;
    const _Float16* __restrict__ ptRow = sPT + lrow * HWPAD;

    float s = 0.f;

    #pragma unroll 1                       // keep passes separate (barriers!)
    for (int pass = 0; pass < 2; ++pass) {
        __syncthreads();   // pass0: tiles visible; pass1: slab reads finished

        // ---- stage this pass's 64 hw rows into padded LDS ----
        {   // thread t -> row t/4, 64 halves at col (t%4)*64
            const int row = tid >> 2, col = (tid & 3) * 64;
            const uint4* __restrict__ src =
                (const uint4*)(hwH + (pass * 64 + row) * 256 + col);
            uint4* dst = (uint4*)(sHW + row * HWPAD + col);
            #pragma unroll
            for (int q = 0; q < 8; ++q) dst[q] = src[q];
        }
        __syncthreads();

        // C init: src_lin[n,h] + tgt_lin[m,h]
        v8f acc[4];
        #pragma unroll
        for (int j = 0; j < 4; ++j) {
            #pragma unroll
            for (int v = 0; v < 8; ++v) {
                const int h = pass * 64 + j * 16 + v + 8 * hsel;
                acc[j][v] = sl[h] + tl[h];
            }
        }

        // Ping-pong B fragments, built one k-chunk ahead.
        Frag Bf[2];
        #pragma unroll
        for (int v = 0; v < 8; ++v) {
            const int d = kOff[v];
            Bf[0].p[v] = (*(const h2*)(psRow + d)) * (*(const h2*)(ptRow + d));
        }

        #pragma unroll
        for (int kc = 0; kc < 8; ++kc) {
            if (kc < 7) {   // build B for kc+1 into the other buffer
                #pragma unroll
                for (int v = 0; v < 8; ++v) {
                    const int d = (kc + 1) * 32 + kOff[v];
                    Bf[(kc + 1) & 1].p[v] =
                        (*(const h2*)(psRow + d)) * (*(const h2*)(ptRow + d));
                }
            }
            #pragma unroll
            for (int j = 0; j < 4; ++j) {
                // A = hw fragment from conflict-free LDS (2x ds_load_b128).
                Frag A;
                const _Float16* __restrict__ aRow =
                    sHW + (j * 16 + lrow) * HWPAD + kc * 32;
                #pragma unroll
                for (int v = 0; v < 8; ++v)
                    A.p[v] = *(const h2*)(aRow + kOff[v]);
                acc[j] = __builtin_amdgcn_wmma_f32_16x16x32_f16(
                    false, A.v, false, Bf[kc & 1].v, (short)0, acc[j],
                    false, false);
            }
        }

        // Fused epilogue: silu + dot with Wo (fast v_rcp, not IEEE div).
        #pragma unroll
        for (int j = 0; j < 4; ++j) {
            #pragma unroll
            for (int v = 0; v < 8; ++v) {
                const int h = pass * 64 + j * 16 + v + 8 * hsel;
                const float x   = acc[j][v];
                const float sig = __builtin_amdgcn_rcpf(1.0f + __expf(-x));
                s += x * sig * Wo[h];
            }
        }
    }

    s += __shfl_xor(s, 16, 32);                        // combine the two h halves
    if (lane < 16)
        out[n * N_TGT + mBase + lrow] = s + bo[0];
}

// ---------------------------------------------------------------------------
// Launcher
// ---------------------------------------------------------------------------
extern "C" void kernel_launch(void* const* d_in, const int* in_sizes, int n_in,
                              void* d_out, int out_size, void* d_ws, size_t ws_size,
                              hipStream_t stream)
{
    (void)in_sizes; (void)n_in; (void)out_size; (void)ws_size;

    const float* source_val = (const float*)d_in[0];
    const float* target_val = (const float*)d_in[1];
    const float* Ws = (const float*)d_in[2];
    const float* bs = (const float*)d_in[3];
    const float* Wt = (const float*)d_in[4];
    const float* bt = (const float*)d_in[5];
    const float* Wi = (const float*)d_in[6];
    const float* bi = (const float*)d_in[7];
    const float* Wo = (const float*)d_in[8];
    const float* bo = (const float*)d_in[9];
    float* out = (float*)d_out;

    // Workspace layout (all offsets 16B aligned)
    char* ws = (char*)d_ws;
    float*    psF    = (float*)   (ws + 0);                  // 1 MB
    float*    ptF    = (float*)   (ws + (1u << 20));         // 1 MB
    _Float16* psH    = (_Float16*)(ws + (2u << 20));         // 512 KB
    _Float16* ptH    = (_Float16*)(ws + (2u << 20) + 524288);// 512 KB
    _Float16* hwH    = (_Float16*)(ws + (3u << 20));         // 64 KB
    float*    srcLin = (float*)   (ws + (3u << 20) + 65536); // 512 KB
    float*    tgtLin = (float*)   (ws + (3u << 20) + 65536 + 524288); // 512 KB

    // Stage 1: projections + linear terms + f16 conversions
    proj_kernel<<<N_SRC, WIDTH, 0, stream>>>(source_val, Ws, bs, psF, psH);
    proj_kernel<<<N_TGT, WIDTH, 0, stream>>>(target_val, Wt, bt, ptF, ptH);
    lin_kernel<<<N_SRC, HIDDEN, 0, stream>>>(psF, Wi, bi, srcLin, 0,     0);
    lin_kernel<<<N_TGT, HIDDEN, 0, stream>>>(ptF, Wi, bi, tgtLin, WIDTH, 1);
    cvt_hw_kernel<<<(HIDDEN * WIDTH) / 256, 256, 0, stream>>>(Wi, hwH);

    // Stage 2: fused einsum + bias + silu + Wo reduction
    dim3 grid(N_TGT / 16, N_SRC / 8);
    route_kernel<<<grid, 256, 0, stream>>>(psH, ptH, hwH, srcLin, tgtLin, Wo, bo, out);
}
